// QGAT_28424093565783
// MI455X (gfx1250) — matrix operand order repaired and checked by reading, not compiled, not run on hardware
//
#include <hip/hip_runtime.h>
#include <hip/hip_bf16.h>
#include <math.h>

#define NN 50000
#define EE 400000
#define EP 450000   /* EE + NN self loops */

typedef __bf16 bf16_t;
typedef bf16_t v16bf __attribute__((ext_vector_type(16)));
typedef float  v8f   __attribute__((ext_vector_type(8)));

union Frag { v16bf v; uint4 q[2]; unsigned short u[16]; };
union Acc8 { v8f v; float f[8]; };

__device__ __forceinline__ unsigned short f2bf(float f) {
  unsigned int u = __float_as_uint(f);
  unsigned int r = u + 0x7FFFu + ((u >> 16) & 1u);   // round-to-nearest-even
  return (unsigned short)(r >> 16);
}

__device__ __forceinline__ void edge_sd(const int* ei, int e, int& s, int& d) {
  if (e < EE) { s = ei[e]; d = ei[EE + e]; }
  else        { s = e - EE; d = e - EE; }
}

__device__ __forceinline__ void atomicMaxF(float* addr, float v) {
  if (v >= 0.0f) atomicMax((int*)addr, __float_as_int(v));
  else           atomicMin((unsigned int*)addr, __float_as_uint(v));
}

/* ---------------- utility fills / casts ---------------- */

__global__ __launch_bounds__(256) void fillf_k(float* p, float v, int n) {
  int i = blockIdx.x * 256 + threadIdx.x;
  if (i < n) p[i] = v;
}

__global__ __launch_bounds__(256) void cast_x_k(const float* x, unsigned short* xbf, int n) {
  int i = blockIdx.x * 256 + threadIdx.x;
  if (i < n) xbf[i] = f2bf(x[i]);
}

/* Pack fused W1 (256 x 512 = Wl1|Wr1) into B-fragment layout:
   frag(kk,nb): 32 lanes x 16 bf16 contiguous per lane.
   lane l holds col n = nb*16 + (l&15), K = kk*32 + (l>>4)*16 + e  */
__global__ __launch_bounds__(256) void packW1_k(const float* Wl, const float* Wr, unsigned short* Wp) {
  int id = blockIdx.x * 256 + threadIdx.x;            // 8*32*32*16 = 131072
  if (id >= 131072) return;
  int e  = id & 15;
  int l  = (id >> 4) & 31;
  int nb = (id >> 9) & 31;
  int kk = id >> 14;
  int K  = kk * 32 + (l >> 4) * 16 + e;
  int n  = nb * 16 + (l & 15);
  float src = (n < 256) ? Wl[K * 256 + n] : Wr[K * 256 + (n - 256)];
  Wp[id] = f2bf(src);
}

/* Pack fused W2 (256 x 32 = Wl2|Wr2) */
__global__ __launch_bounds__(256) void packW2_k(const float* Wl, const float* Wr, unsigned short* Wp) {
  int id = blockIdx.x * 256 + threadIdx.x;            // 8*2*32*16 = 8192
  if (id >= 8192) return;
  int e  = id & 15;
  int l  = (id >> 4) & 31;
  int nb = (id >> 9) & 1;
  int kk = id >> 10;
  int K  = kk * 32 + (l >> 4) * 16 + e;
  int n  = nb * 16 + (l & 15);
  float src = (n < 16) ? Wl[K * 16 + n] : Wr[K * 16 + (n - 16)];
  Wp[id] = f2bf(src);
}

/* ---------------- GEMM layer 1: [16 x 256] @ [256 x 512] per block ----------------
   8 waves/block; wave w covers cols [w*64, w*64+64) via 4 wmma accumulators. */
__global__ __launch_bounds__(256) void gemm1_k(const unsigned short* __restrict__ xbf,
                                               const unsigned short* __restrict__ Wp,
                                               const float* __restrict__ bl,
                                               const float* __restrict__ br,
                                               float* __restrict__ xl1,
                                               float* __restrict__ xr1) {
  __shared__ __align__(16) unsigned short ldsA[16 * 256];
  int t   = threadIdx.x;
  int blk = blockIdx.x;                       // 3125 row tiles (N = 16*3125)
  const uint4* srcq = (const uint4*)(xbf + (size_t)blk * 16 * 256);
  ((uint4*)ldsA)[t]       = srcq[t];
  ((uint4*)ldsA)[t + 256] = srcq[t + 256];
  __syncthreads();

  int w = t >> 5, l = t & 31;
  int row = l & 15, half = l >> 4;
  int nb0 = w * 4;

  v8f acc[4];
  acc[0] = (v8f){}; acc[1] = (v8f){}; acc[2] = (v8f){}; acc[3] = (v8f){};

  for (int kk = 0; kk < 8; ++kk) {
    int kbase = kk * 32;
    Frag a;
    a.q[0] = *(const uint4*)&ldsA[row * 256 + kbase + half * 16];
    a.q[1] = *(const uint4*)&ldsA[row * 256 + kbase + half * 16 + 8];
    const unsigned short* bp = Wp + ((size_t)(kk * 32 + nb0) * 32 + l) * 16;
#pragma unroll
    for (int jj = 0; jj < 4; ++jj) {
      Frag b;
      b.q[0] = *(const uint4*)(bp + (size_t)jj * 512);
      b.q[1] = *(const uint4*)(bp + (size_t)jj * 512 + 8);
      acc[jj] = __builtin_amdgcn_wmma_f32_16x16x32_bf16(
          false, a.v, false, b.v, (short)0, acc[jj], false, false);
    }
  }

#pragma unroll
  for (int jj = 0; jj < 4; ++jj) {
    Acc8 r; r.v = acc[jj];
    int nc = (nb0 + jj) * 16 + (l & 15);
    float bias = (nc < 256) ? bl[nc] : br[nc - 256];
    float* dst = (nc < 256) ? (xl1 + (size_t)blk * 16 * 256 + nc)
                            : (xr1 + (size_t)blk * 16 * 256 + (nc - 256));
#pragma unroll
    for (int j = 0; j < 8; ++j) {
      int m = j + half * 8;
      dst[(size_t)m * 256] = r.f[j] + bias;
    }
  }
}

/* ---------------- edge attention, layer 1 (H=4, C=64): one wave per edge ---------- */
__global__ __launch_bounds__(256) void edge_att1_k(const int* __restrict__ ei,
                                                   const float* __restrict__ xl1,
                                                   const float* __restrict__ xr1,
                                                   const float* __restrict__ att,
                                                   float* __restrict__ e1,
                                                   float* __restrict__ emax1) {
  int t = threadIdx.x;
  int e = blockIdx.x * 8 + (t >> 5);
  if (e >= EP) return;
  int l = t & 31;
  int s, d; edge_sd(ei, e, s, d);
  int v0 = l * 8;
  const float4* pl = (const float4*)(xl1 + (size_t)s * 256 + v0);
  const float4* pr = (const float4*)(xr1 + (size_t)d * 256 + v0);
  const float4* pa = (const float4*)(att + v0);
  float sum = 0.0f;
#pragma unroll
  for (int q = 0; q < 2; ++q) {
    float4 a = pl[q], b = pr[q], c = pa[q];
    float m;
    m = a.x + b.x; m = (m > 0.f) ? m : 0.2f * m; sum += m * c.x;
    m = a.y + b.y; m = (m > 0.f) ? m : 0.2f * m; sum += m * c.y;
    m = a.z + b.z; m = (m > 0.f) ? m : 0.2f * m; sum += m * c.z;
    m = a.w + b.w; m = (m > 0.f) ? m : 0.2f * m; sum += m * c.w;
  }
  sum += __shfl_xor(sum, 1);
  sum += __shfl_xor(sum, 2);
  sum += __shfl_xor(sum, 4);
  if ((l & 7) == 0) {
    int h = l >> 3;
    e1[(size_t)e * 4 + h] = sum;
    atomicMaxF(&emax1[(size_t)d * 4 + h], sum);
  }
}

__global__ __launch_bounds__(256) void edge_exp1_k(const int* __restrict__ ei,
                                                   const float* __restrict__ e1,
                                                   const float* __restrict__ emax1,
                                                   float* __restrict__ ee1,
                                                   float* __restrict__ den1) {
  int id = blockIdx.x * 256 + threadIdx.x;           // EP*4
  if (id >= EP * 4) return;
  int e = id >> 2, h = id & 3;
  int s, d; edge_sd(ei, e, s, d); (void)s;
  float ee = __expf(e1[id] - emax1[(size_t)d * 4 + h]);
  ee1[id] = ee;
  atomicAdd(&den1[(size_t)d * 4 + h], ee);
}

__global__ __launch_bounds__(256) void edge_agg1_k(const int* __restrict__ ei,
                                                   const float* __restrict__ xl1,
                                                   const float* __restrict__ ee1,
                                                   const float* __restrict__ den1,
                                                   float* __restrict__ out1) {
  int t = threadIdx.x;
  int e = blockIdx.x * 8 + (t >> 5);
  if (e >= EP) return;
  int l = t & 31;
  int s, d; edge_sd(ei, e, s, d);
  int h = l >> 3;
  float alpha = ee1[(size_t)e * 4 + h] / (den1[(size_t)d * 4 + h] + 1e-16f);
  int v0 = l * 8;
  const float4* pl = (const float4*)(xl1 + (size_t)s * 256 + v0);
  float* po = out1 + (size_t)d * 256 + v0;
#pragma unroll
  for (int q = 0; q < 2; ++q) {
    float4 a = pl[q];
    atomicAdd(po + q * 4 + 0, alpha * a.x);
    atomicAdd(po + q * 4 + 1, alpha * a.y);
    atomicAdd(po + q * 4 + 2, alpha * a.z);
    atomicAdd(po + q * 4 + 3, alpha * a.w);
  }
}

/* tanh(out1 + bias1) -> bf16 h */
__global__ __launch_bounds__(256) void post1_k(const float* __restrict__ out1,
                                               const float* __restrict__ bias1,
                                               unsigned short* __restrict__ hbf) {
  int id = blockIdx.x * 256 + threadIdx.x;
  if (id >= NN * 256) return;
  int c = id & 255;
  hbf[id] = f2bf(tanhf(out1[id] + bias1[c]));
}

/* ---------------- GEMM layer 2: wave per 16-row tile, 32 fused cols -------------- */
__global__ __launch_bounds__(256) void gemm2_k(const unsigned short* __restrict__ hbf,
                                               const unsigned short* __restrict__ Wp,
                                               const float* __restrict__ bl,
                                               const float* __restrict__ br,
                                               float* __restrict__ xl2,
                                               float* __restrict__ xr2) {
  int t = threadIdx.x;
  int g = blockIdx.x * 8 + (t >> 5);                 // 3125 row tiles
  if (g >= 3125) return;
  int l = t & 31;
  int row = g * 16 + (l & 15);
  int half = l >> 4;

  v8f acc[2];
  acc[0] = (v8f){}; acc[1] = (v8f){};

  for (int kk = 0; kk < 8; ++kk) {
    int kbase = kk * 32;
    Frag a;
    const unsigned short* ap = hbf + (size_t)row * 256 + kbase + half * 16;
    a.q[0] = *(const uint4*)ap;
    a.q[1] = *(const uint4*)(ap + 8);
#pragma unroll
    for (int nb = 0; nb < 2; ++nb) {
      Frag b;
      const unsigned short* bp = Wp + ((size_t)(kk * 2 + nb) * 32 + l) * 16;
      b.q[0] = *(const uint4*)bp;
      b.q[1] = *(const uint4*)(bp + 8);
      acc[nb] = __builtin_amdgcn_wmma_f32_16x16x32_bf16(
          false, a.v, false, b.v, (short)0, acc[nb], false, false);
    }
  }

#pragma unroll
  for (int nb = 0; nb < 2; ++nb) {
    Acc8 r; r.v = acc[nb];
    int nc = nb * 16 + (l & 15);
    float bias = (nc < 16) ? bl[nc] : br[nc - 16];
    float* dst = (nc < 16) ? (xl2 + (size_t)g * 16 * 16 + nc)
                           : (xr2 + (size_t)g * 16 * 16 + (nc - 16));
#pragma unroll
    for (int j = 0; j < 8; ++j) {
      int m = j + half * 8;
      dst[(size_t)m * 16] = r.f[j] + bias;
    }
  }
}

/* ---------------- edge attention, layer 2 (H=1, C=16): thread per edge ----------- */
__global__ __launch_bounds__(256) void edge_att2_k(const int* __restrict__ ei,
                                                   const float* __restrict__ xl2,
                                                   const float* __restrict__ xr2,
                                                   const float* __restrict__ att,
                                                   float* __restrict__ e2,
                                                   float* __restrict__ emax2) {
  int e = blockIdx.x * 256 + threadIdx.x;
  if (e >= EP) return;
  int s, d; edge_sd(ei, e, s, d);
  const float4* pl = (const float4*)(xl2 + (size_t)s * 16);
  const float4* pr = (const float4*)(xr2 + (size_t)d * 16);
  const float4* pa = (const float4*)att;
  float sum = 0.0f;
#pragma unroll
  for (int q = 0; q < 4; ++q) {
    float4 a = pl[q], b = pr[q], c = pa[q];
    float m;
    m = a.x + b.x; m = (m > 0.f) ? m : 0.2f * m; sum += m * c.x;
    m = a.y + b.y; m = (m > 0.f) ? m : 0.2f * m; sum += m * c.y;
    m = a.z + b.z; m = (m > 0.f) ? m : 0.2f * m; sum += m * c.z;
    m = a.w + b.w; m = (m > 0.f) ? m : 0.2f * m; sum += m * c.w;
  }
  e2[e] = sum;
  atomicMaxF(&emax2[d], sum);
}

__global__ __launch_bounds__(256) void edge_exp2_k(const int* __restrict__ ei,
                                                   const float* __restrict__ e2,
                                                   const float* __restrict__ emax2,
                                                   float* __restrict__ ee2,
                                                   float* __restrict__ den2) {
  int e = blockIdx.x * 256 + threadIdx.x;
  if (e >= EP) return;
  int s, d; edge_sd(ei, e, s, d); (void)s;
  float ee = __expf(e2[e] - emax2[d]);
  ee2[e] = ee;
  atomicAdd(&den2[d], ee);
}

__global__ __launch_bounds__(256) void edge_agg2_k(const int* __restrict__ ei,
                                                   const float* __restrict__ xl2,
                                                   const float* __restrict__ ee2,
                                                   const float* __restrict__ den2,
                                                   float* __restrict__ out2) {
  int id = blockIdx.x * 256 + threadIdx.x;           // EP*16
  if (id >= EP * 16) return;
  int e = id >> 4, c = id & 15;
  int s, d; edge_sd(ei, e, s, d);
  float alpha = ee2[e] / (den2[d] + 1e-16f);
  atomicAdd(&out2[(size_t)d * 16 + c], alpha * xl2[(size_t)s * 16 + c]);
}

/* ---------------- final: o = out2 + bias2; log_softmax ---------------- */
__global__ __launch_bounds__(256) void final_k(const float* __restrict__ out2,
                                               const float* __restrict__ bias2,
                                               float* __restrict__ out) {
  int n = blockIdx.x * 256 + threadIdx.x;
  if (n >= NN) return;
  float o[16];
  float mx = -INFINITY;
#pragma unroll
  for (int c = 0; c < 16; ++c) {
    o[c] = out2[(size_t)n * 16 + c] + bias2[c];
    mx = fmaxf(mx, o[c]);
  }
  float sum = 0.0f;
#pragma unroll
  for (int c = 0; c < 16; ++c) sum += __expf(o[c] - mx);
  float ls = logf(sum);
#pragma unroll
  for (int c = 0; c < 16; ++c) {
    out[(size_t)n * 16 + c] = o[c];
    out[(size_t)NN * 16 + (size_t)n * 16 + c] = o[c] - mx - ls;
  }
}

/* ==================================================================== */

static inline int cdiv(long long a, long long b) { return (int)((a + b - 1) / b); }

extern "C" void kernel_launch(void* const* d_in, const int* in_sizes, int n_in,
                              void* d_out, int out_size, void* d_ws, size_t ws_size,
                              hipStream_t stream) {
  const float* x     = (const float*)d_in[0];
  const int*   ei    = (const int*)d_in[1];
  const float* Wl1   = (const float*)d_in[2];
  const float* bl1   = (const float*)d_in[3];
  const float* Wr1   = (const float*)d_in[4];
  const float* br1   = (const float*)d_in[5];
  const float* att1  = (const float*)d_in[6];
  const float* bias1 = (const float*)d_in[7];
  const float* Wl2   = (const float*)d_in[8];
  const float* bl2   = (const float*)d_in[9];
  const float* Wr2   = (const float*)d_in[10];
  const float* br2   = (const float*)d_in[11];
  const float* att2  = (const float*)d_in[12];
  const float* bias2 = (const float*)d_in[13];
  float* out = (float*)d_out;

  /* carve workspace (256B aligned regions) */
  size_t off = 0;
  auto carve = [&](size_t bytes) -> char* {
    char* p = (char*)d_ws + off;
    off += (bytes + 255) & ~(size_t)255;
    return p;
  };
  unsigned short* xbf  = (unsigned short*)carve((size_t)NN * 256 * 2);
  unsigned short* Wp1  = (unsigned short*)carve((size_t)131072 * 2);
  unsigned short* Wp2  = (unsigned short*)carve((size_t)8192 * 2);
  float* xl1   = (float*)carve((size_t)NN * 256 * 4);
  float* xr1   = (float*)carve((size_t)NN * 256 * 4);
  float* out1  = (float*)carve((size_t)NN * 256 * 4);
  float* e1    = (float*)carve((size_t)EP * 4 * 4);
  float* ee1   = (float*)carve((size_t)EP * 4 * 4);
  float* emax1 = (float*)carve((size_t)NN * 4 * 4);
  float* den1  = (float*)carve((size_t)NN * 4 * 4);
  unsigned short* hbf = (unsigned short*)carve((size_t)NN * 256 * 2);
  float* xl2   = (float*)carve((size_t)NN * 16 * 4);
  float* xr2   = (float*)carve((size_t)NN * 16 * 4);
  float* e2    = (float*)carve((size_t)EP * 4);
  float* ee2   = (float*)carve((size_t)EP * 4);
  float* emax2 = (float*)carve((size_t)NN * 4);
  float* den2  = (float*)carve((size_t)NN * 4);
  float* out2  = (float*)carve((size_t)NN * 16 * 4);
  (void)ws_size; (void)n_in; (void)in_sizes; (void)out_size;

  /* per-call init of accumulation buffers */
  fillf_k<<<cdiv((long long)NN * 256, 256), 256, 0, stream>>>(out1, 0.0f, NN * 256);
  fillf_k<<<cdiv(NN * 4, 256), 256, 0, stream>>>(emax1, -INFINITY, NN * 4);
  fillf_k<<<cdiv(NN * 4, 256), 256, 0, stream>>>(den1, 0.0f, NN * 4);
  fillf_k<<<cdiv(NN * 16, 256), 256, 0, stream>>>(out2, 0.0f, NN * 16);
  fillf_k<<<cdiv(NN, 256), 256, 0, stream>>>(emax2, -INFINITY, NN);
  fillf_k<<<cdiv(NN, 256), 256, 0, stream>>>(den2, 0.0f, NN);

  /* prep: casts + weight packing */
  cast_x_k<<<cdiv((long long)NN * 256, 256), 256, 0, stream>>>(x, xbf, NN * 256);
  packW1_k<<<cdiv(131072, 256), 256, 0, stream>>>(Wl1, Wr1, Wp1);
  packW2_k<<<cdiv(8192, 256), 256, 0, stream>>>(Wl2, Wr2, Wp2);

  /* layer 1 */
  gemm1_k<<<3125, 256, 0, stream>>>(xbf, Wp1, bl1, br1, xl1, xr1);
  edge_att1_k<<<cdiv(EP, 8), 256, 0, stream>>>(ei, xl1, xr1, att1, e1, emax1);
  edge_exp1_k<<<cdiv((long long)EP * 4, 256), 256, 0, stream>>>(ei, e1, emax1, ee1, den1);
  edge_agg1_k<<<cdiv(EP, 8), 256, 0, stream>>>(ei, xl1, ee1, den1, out1);
  post1_k<<<cdiv((long long)NN * 256, 256), 256, 0, stream>>>(out1, bias1, hbf);

  /* layer 2 */
  gemm2_k<<<cdiv(3125, 8), 256, 0, stream>>>(hbf, Wp2, bl2, br2, xl2, xr2);
  edge_att2_k<<<cdiv(EP, 256), 256, 0, stream>>>(ei, xl2, xr2, att2, e2, emax2);
  edge_exp2_k<<<cdiv(EP, 256), 256, 0, stream>>>(ei, e2, emax2, ee2, den2);
  edge_agg2_k<<<cdiv((long long)EP * 16, 256), 256, 0, stream>>>(ei, xl2, ee2, den2, out2);

  /* output */
  final_k<<<cdiv(NN, 256), 256, 0, stream>>>(out2, bias2, out);
}
